// GATAttackPredictor_64570538328560
// MI455X (gfx1250) — matrix operand order repaired and checked by reading, not compiled, not run on hardware
//
#include <hip/hip_runtime.h>
#include <hip/hip_bf16.h>

// ---------------------------------------------------------------------------
// GAT (3-layer) for MI455X / gfx1250.
//  - GEMMs via v_wmma_f32_16x16x32_bf16 with split-bf16 (hi+lo) correction.
//  - Edge softmax via order-preserving u32 atomic max + f32 atomic adds (L2).
//  - attn written in place into d_out; no per-edge scratch tensors.
// ---------------------------------------------------------------------------

typedef __attribute__((ext_vector_type(16))) __bf16 v16bf;
typedef __attribute__((ext_vector_type(8)))  float  v8f;

#define CDIV(a, b) (((a) + (b) - 1) / (b))

// ---------------- WMMA GEMM: H[M,Nout] = X[M,K] @ W[K,Nout] (fp32 in/out) --
__global__ void __launch_bounds__(256)
gat_gemm_wmma(const float* __restrict__ X, const float* __restrict__ W,
              float* __restrict__ H, int M, int K, int Nout) {
  const int wave   = threadIdx.x >> 5;
  const int lane   = threadIdx.x & 31;
  const int tilesN = Nout >> 4;
  const int tile   = blockIdx.x * 8 + wave;
  const int nTiles = (M >> 4) * tilesN;
  if (tile >= nTiles) return;                 // wave-uniform: EXEC stays all-1s
  const int m0 = (tile / tilesN) << 4;
  const int n0 = (tile % tilesN) << 4;
  const int mA = lane & 15;                   // A: row this lane holds
  const int hA = lane >> 4;                   // A: K-half selector

  v8f acc = {};
  for (int k0 = 0; k0 < K; k0 += 32) {
    v16bf ah, al, bh, bl;
    // ---- A fragment: 16x32, lane m=L&15, VGPR v -> K pair --------------
    const float* xr = X + (size_t)(m0 + mA) * K + k0;
#pragma unroll
    for (int v = 0; v < 8; ++v) {
      const int kk = ((v >> 2) << 4) + (hA << 3) + ((v & 3) << 1);
      const float f0 = xr[kk], f1 = xr[kk + 1];
      const __bf16 h0 = (__bf16)f0, h1 = (__bf16)f1;
      ah[2 * v] = h0;                 ah[2 * v + 1] = h1;
      al[2 * v] = (__bf16)(f0 - (float)h0);
      al[2 * v + 1] = (__bf16)(f1 - (float)h1);
    }
    // ---- B fragment: 32x16, lane n=L&15, VGPR v -> K pair (same as A) ---
    const float* wr = W + (size_t)k0 * Nout + n0 + mA;
#pragma unroll
    for (int v = 0; v < 8; ++v) {
      const int kk = ((v >> 2) << 4) + (hA << 3) + ((v & 3) << 1);
      const float f0 = wr[(size_t)kk * Nout], f1 = wr[(size_t)(kk + 1) * Nout];
      const __bf16 h0 = (__bf16)f0, h1 = (__bf16)f1;
      bh[2 * v] = h0;                 bh[2 * v + 1] = h1;
      bl[2 * v] = (__bf16)(f0 - (float)h0);
      bl[2 * v + 1] = (__bf16)(f1 - (float)h1);
    }
    // split-bf16: a*b ~= ah*bh + ah*bl + al*bh (al*bl below fp32 noise)
    acc = __builtin_amdgcn_wmma_f32_16x16x32_bf16(false, al, false, bh, (short)0, acc, false, false);
    acc = __builtin_amdgcn_wmma_f32_16x16x32_bf16(false, ah, false, bl, (short)0, acc, false, false);
    acc = __builtin_amdgcn_wmma_f32_16x16x32_bf16(false, ah, false, bh, (short)0, acc, false, false);
  }
  // ---- D store: lane = n + 16*(m/8), VGPR r = m%8 ----------------------
  const int nD = lane & 15;
  const int mD = (lane >> 4) << 3;
  float* orow = H + (size_t)(m0 + mD) * Nout + n0 + nD;
#pragma unroll
  for (int r = 0; r < 8; ++r) orow[(size_t)r * Nout] = acc[r];
}

// ---------------- per-(node,head) attention scalars ------------------------
__global__ void gat_alpha(const float* __restrict__ H,
                          const float* __restrict__ a_src,
                          const float* __restrict__ a_dst,
                          float* __restrict__ as_out, float* __restrict__ ad_out,
                          int Nn, int heads, int C) {
  const int i = blockIdx.x * blockDim.x + threadIdx.x;
  if (i >= Nn * heads) return;
  const int hd = i % heads;
  const float* hp = H + (size_t)(i / heads) * heads * C + (size_t)hd * C;
  const float* pa = a_src + (size_t)hd * C;
  const float* pb = a_dst + (size_t)hd * C;
  float s1 = 0.f, s2 = 0.f;
  for (int c = 0; c < C; ++c) { const float v = hp[c]; s1 += v * pa[c]; s2 += v * pb[c]; }
  as_out[i] = s1;
  ad_out[i] = s2;
}

// ---------------- order-preserving float<->u32 for atomic max --------------
__device__ __forceinline__ unsigned fenc(float f) {
  const unsigned u = __float_as_uint(f);
  return (u & 0x80000000u) ? ~u : (u | 0x80000000u);
}
__device__ __forceinline__ float fdec(unsigned u) {
  return (u & 0x80000000u) ? __uint_as_float(u & 0x7fffffffu) : __uint_as_float(~u);
}
#define FENC_NEG_INF 0x007FFFFFu  // fenc(-inf)

__global__ void fill_u32(unsigned* __restrict__ p, unsigned v, long long n) {
  const long long i = (long long)blockIdx.x * blockDim.x + threadIdx.x;
  if (i < n) p[i] = v;
}

__device__ __forceinline__ void edge_nodes(int e, int E_, const int* src, const int* dst,
                                           int& s, int& d) {
  if (e < E_) { s = src[e]; d = dst[e]; }
  else        { s = e - E_; d = e - E_; }   // self-loops appended
}

// ---------------- edge pass 1: segment max (leaky-relu logits) -------------
__global__ void gat_edge_max(const int* __restrict__ src, const int* __restrict__ dst,
                             const float* __restrict__ as, const float* __restrict__ ad,
                             unsigned* __restrict__ menc, int E_, int Nn, int heads) {
  const int i = blockIdx.x * blockDim.x + threadIdx.x;
  if (i >= (E_ + Nn) * heads) return;
  const int hd = i % heads;
  int s, d; edge_nodes(i / heads, E_, src, dst, s, d);
  float x = as[s * heads + hd] + ad[d * heads + hd];
  x = x > 0.f ? x : 0.2f * x;
  atomicMax(&menc[d * heads + hd], fenc(x));
}

// ---------------- edge pass 2: ex = exp(e - m), segment sum ----------------
__global__ void gat_edge_exp(const int* __restrict__ src, const int* __restrict__ dst,
                             const float* __restrict__ as, const float* __restrict__ ad,
                             const unsigned* __restrict__ menc, float* __restrict__ ssum,
                             float* __restrict__ attn, int E_, int Nn, int heads) {
  const int i = blockIdx.x * blockDim.x + threadIdx.x;
  if (i >= (E_ + Nn) * heads) return;
  const int hd = i % heads;
  int s, d; edge_nodes(i / heads, E_, src, dst, s, d);
  float x = as[s * heads + hd] + ad[d * heads + hd];
  x = x > 0.f ? x : 0.2f * x;
  const float ex = __expf(x - fdec(menc[d * heads + hd]));
  attn[i] = ex;
  atomicAdd(&ssum[d * heads + hd], ex);
}

// ---------------- edge pass 3: normalize alpha in place --------------------
__global__ void gat_edge_norm(const int* __restrict__ src, const int* __restrict__ dst,
                              const float* __restrict__ ssum, float* __restrict__ attn,
                              int E_, int Nn, int heads) {
  const int i = blockIdx.x * blockDim.x + threadIdx.x;
  if (i >= (E_ + Nn) * heads) return;
  const int hd = i % heads;
  int s, d; edge_nodes(i / heads, E_, src, dst, s, d);
  attn[i] = attn[i] / (ssum[d * heads + hd] + 1e-16f);
}

// ---------------- edge pass 4: scatter messages (f32 atomics, L2) ----------
__global__ void gat_edge_msg(const int* __restrict__ src, const int* __restrict__ dst,
                             const float* __restrict__ H, const float* __restrict__ attn,
                             float* __restrict__ acc, int E_, int Nn, int heads, int C) {
  const int tpe = (heads * C) >> 2;                 // threads per edge (float4)
  const int gid = blockIdx.x * blockDim.x + threadIdx.x;
  const int e = gid / tpe;
  if (e >= E_ + Nn) return;
  const int r  = gid % tpe;
  const int hd = r / (C >> 2);
  const int c4 = (r % (C >> 2)) << 2;
  int s, d; edge_nodes(e, E_, src, dst, s, d);
  const float alpha = attn[(size_t)e * heads + hd];
  const float4 hv = *(const float4*)(H + ((size_t)s * heads + hd) * C + c4);
  float* ap = acc + ((size_t)d * heads + hd) * C + c4;
  atomicAdd(ap + 0, hv.x * alpha);
  atomicAdd(ap + 1, hv.y * alpha);
  atomicAdd(ap + 2, hv.z * alpha);
  atomicAdd(ap + 3, hv.w * alpha);
}

// ---------------- bias + optional ELU --------------------------------------
__global__ void gat_bias_act(const float* __restrict__ acc, const float* __restrict__ b,
                             float* __restrict__ out, long long total, int F, int do_elu) {
  const long long i = (long long)blockIdx.x * blockDim.x + threadIdx.x;
  if (i >= total) return;
  float v = acc[i] + b[i % F];
  if (do_elu) v = v > 0.f ? v : (__expf(v) - 1.f);
  out[i] = v;
}

// ---------------------------------------------------------------------------
static void run_layer(const float* Xin, int Fin, const float* Wt,
                      const float* a_s, const float* a_d, int heads, int C,
                      const int* src, const int* dst, int E_, int Nn,
                      float* Hbuf, float* asb, float* adb, unsigned* menc,
                      float* ssum, float* attn, float* accbuf, hipStream_t stream) {
  const int Fout = heads * C;
  const int Et   = E_ + Nn;
  // 1) H = X @ W  (WMMA; one wave per 16x16 tile, 8 waves per block)
  {
    const int tiles = (Nn >> 4) * (Fout >> 4);
    gat_gemm_wmma<<<CDIV(tiles, 8), 256, 0, stream>>>(Xin, Wt, Hbuf, Nn, Fin, Fout);
  }
  // 2) per-node attention scalars
  gat_alpha<<<CDIV(Nn * heads, 256), 256, 0, stream>>>(Hbuf, a_s, a_d, asb, adb, Nn, heads, C);
  // 3) init segment buffers + output accumulator (stream-ordered after GEMM,
  //    so accbuf may alias the GEMM input buffer)
  fill_u32<<<CDIV(Nn * heads, 256), 256, 0, stream>>>(menc, FENC_NEG_INF, (long long)Nn * heads);
  fill_u32<<<CDIV(Nn * heads, 256), 256, 0, stream>>>((unsigned*)ssum, 0u, (long long)Nn * heads);
  fill_u32<<<CDIV((long long)Nn * Fout, 256), 256, 0, stream>>>((unsigned*)accbuf, 0u, (long long)Nn * Fout);
  // 4-6) segment softmax
  const int eh = Et * heads;
  gat_edge_max <<<CDIV(eh, 256), 256, 0, stream>>>(src, dst, asb, adb, menc, E_, Nn, heads);
  gat_edge_exp <<<CDIV(eh, 256), 256, 0, stream>>>(src, dst, asb, adb, menc, ssum, attn, E_, Nn, heads);
  gat_edge_norm<<<CDIV(eh, 256), 256, 0, stream>>>(src, dst, ssum, attn, E_, Nn, heads);
  // 7) weighted message scatter
  const int mt = Et * ((heads * C) >> 2);
  gat_edge_msg<<<CDIV(mt, 256), 256, 0, stream>>>(src, dst, Hbuf, attn, accbuf, E_, Nn, heads, C);
}

extern "C" void kernel_launch(void* const* d_in, const int* in_sizes, int n_in,
                              void* d_out, int out_size, void* d_ws, size_t ws_size,
                              hipStream_t stream) {
  (void)n_in; (void)out_size; (void)ws_size;
  const float* x    = (const float*)d_in[0];
  const int*   eix  = (const int*)d_in[1];
  const float* W1   = (const float*)d_in[2];
  const float* as1  = (const float*)d_in[3];
  const float* ad1  = (const float*)d_in[4];
  const float* b1   = (const float*)d_in[5];
  const float* W2   = (const float*)d_in[6];
  const float* as2  = (const float*)d_in[7];
  const float* ad2  = (const float*)d_in[8];
  const float* b2   = (const float*)d_in[9];
  const float* W3   = (const float*)d_in[10];
  const float* as3  = (const float*)d_in[11];
  const float* ad3  = (const float*)d_in[12];
  const float* b3   = (const float*)d_in[13];

  const int N  = in_sizes[0] / 128;       // 50000
  const int E_ = in_sizes[1] / 2;         // 1600000
  const int Et = E_ + N;
  const int* src = eix;
  const int* dst = eix + E_;

  // d_out layout: out[N*64] | attn1[Et*8] | attn2[Et*8] | attn3[Et]
  float* out_p  = (float*)d_out;
  float* attn1  = out_p + (size_t)N * 64;
  float* attn2  = attn1 + (size_t)Et * 8;
  float* attn3  = attn2 + (size_t)Et * 8;

  // workspace
  float* wsf = (float*)d_ws;
  size_t off = 0;
  float*    bufA = wsf + off; off += (size_t)N * 256;   // pre-attn features H
  float*    bufB = wsf + off; off += (size_t)N * 256;   // layer input / accumulator
  float*    asb  = wsf + off; off += (size_t)N * 8;
  float*    adb  = wsf + off; off += (size_t)N * 8;
  unsigned* menc = (unsigned*)(wsf + off); off += (size_t)N * 8;
  float*    ssum = wsf + off; off += (size_t)N * 8;

  // ---- layer 1: x[N,128] -> h1[N,8,32], attn1 ----
  run_layer(x, 128, W1, as1, ad1, 8, 32, src, dst, E_, N,
            bufA, asb, adb, menc, ssum, attn1, bufB, stream);
  gat_bias_act<<<CDIV((long long)N * 256, 256), 256, 0, stream>>>(bufB, b1, bufB, (long long)N * 256, 256, 1);

  // ---- layer 2: h[N,256] -> h2[N,8,32], attn2 (acc reuses bufB after GEMM) ----
  run_layer(bufB, 256, W2, as2, ad2, 8, 32, src, dst, E_, N,
            bufA, asb, adb, menc, ssum, attn2, bufB, stream);
  gat_bias_act<<<CDIV((long long)N * 256, 256), 256, 0, stream>>>(bufB, b2, bufB, (long long)N * 256, 256, 1);

  // ---- layer 3: h[N,256] -> out[N,64] (heads=1, mean == identity), attn3 ----
  run_layer(bufB, 256, W3, as3, ad3, 1, 64, src, dst, E_, N,
            bufA, asb, adb, menc, ssum, attn3, bufB, stream);
  gat_bias_act<<<CDIV((long long)N * 64, 256), 256, 0, stream>>>(bufB, b3, out_p, (long long)N * 64, 64, 0);
}
